// MultiHeadSelfAttention_70222715289849
// MI455X (gfx1250) — matrix operand order repaired and checked by reading, not compiled, not run on hardware
//
#include <hip/hip_runtime.h>

#define DEV __device__ __forceinline__

typedef __attribute__((ext_vector_type(16))) __bf16 v16bf;
typedef __attribute__((ext_vector_type(8)))  __bf16 v8bf;
typedef __attribute__((ext_vector_type(8)))  float  v8f;

struct BfPair { v8bf lo, hi; };

constexpr int Bc  = 2;
constexpr int Sc  = 2048;
constexpr int Dc  = 1024;
constexpr int Hc  = 16;
constexpr int HDc = 64;
constexpr int SBc = Bc * Sc;  // 4096 flat rows
// log2(e) / sqrt(HD) : fold softmax scale into exp2
constexpr float kC1 = 0.18033688011112042f;

DEV __bf16 f32_to_bf16(float f) {
  unsigned u = __builtin_bit_cast(unsigned, f);
  u += 0x7FFFu + ((u >> 16) & 1u);  // round-to-nearest-even
  unsigned short h = (unsigned short)(u >> 16);
  return __builtin_bit_cast(__bf16, h);
}

DEV v8f wmma_bf16(v16bf a, v16bf b, v8f c) {
  // D = A(16x32) * B(32x16) + C, f32 accumulate
  return __builtin_amdgcn_wmma_f32_16x16x32_bf16(
      /*neg_a=*/false, a, /*neg_b=*/false, b,
      /*c_mod=*/(short)0, c, /*reuse_a=*/false, /*reuse_b=*/false);
}

// A fragment (16x32, lane = M row): lanes<16 hold K = {k0..k0+7, k0+16..k0+23},
// lanes>=16 hold K = {k0+8..k0+15, k0+24..k0+31}. Source row-major, stride ld.
DEV v16bf load_fragA(const __bf16* __restrict__ p, int ld, int k0, int lane) {
  const int row = lane & 15;
  const int kb  = k0 + ((lane >> 4) << 3);
  const __bf16* q = p + (size_t)row * ld + kb;
  BfPair pr{*(const v8bf*)(q), *(const v8bf*)(q + 16)};
  return __builtin_bit_cast(v16bf, pr);
}

// B fragment (32x16, lane = N column): lanes<16 hold K = {k0..k0+7, k0+16..k0+23},
// lanes>=16 hold K = {k0+8..k0+15, k0+24..k0+31}. Source holds B^T row-major (row n = col n
// of B, i.e. "N x K" row-major), stride ld. Two 16B loads.
DEV v16bf load_fragB(const __bf16* __restrict__ p, int ld, int k0, int lane) {
  const int row = lane & 15;
  const int kb  = k0 + ((lane >> 4) << 3);
  const __bf16* q = p + (size_t)row * ld + kb;
  BfPair pr{*(const v8bf*)(q), *(const v8bf*)(q + 16)};
  return __builtin_bit_cast(v16bf, pr);
}

// ---------------------------------------------------------------- fp32->bf16
__global__ void cvt_f32_bf16(const float* __restrict__ in,
                             __bf16* __restrict__ out, int n) {
  int i = blockIdx.x * blockDim.x + threadIdx.x;
  int stride = gridDim.x * blockDim.x;
  for (; i < n; i += stride) out[i] = f32_to_bf16(in[i]);
}

// ---------------------------------------------------------------- WMMA GEMM
// C[m][n] = sum_k X[m][k] * W[n][k] + bias[n]   (M=4096, N=K=1024)
// Wave tile 64(M) x 32(N): 8 WMMAs per 6 fragment loads per k-step.
// MODE 0: out bf16 head-major  Q/K  [B][H][S][HD]
// MODE 1: out bf16 head-transposed V [B][H][HD][S]
// MODE 2: out fp32 row-major [M][N]
template <int MODE>
__global__ __launch_bounds__(256) void gemm_bf16(
    const __bf16* __restrict__ X, const __bf16* __restrict__ W,
    const float* __restrict__ bias, void* __restrict__ outp) {
  const int lane = threadIdx.x & 31;
  const int wid  = threadIdx.x >> 5;
  const int m_base = blockIdx.x * 64;
  const int n_base = blockIdx.y * 256 + wid * 32;

  const __bf16* xa = X + (size_t)m_base * Dc;
  const __bf16* wb = W + (size_t)n_base * Dc;

  v8f c[4][2];
#pragma unroll
  for (int i = 0; i < 4; ++i)
#pragma unroll
    for (int j = 0; j < 2; ++j) c[i][j] = (v8f){};

#pragma unroll 2
  for (int k0 = 0; k0 < Dc; k0 += 32) {
    v16bf a[4], b[2];
#pragma unroll
    for (int i = 0; i < 4; ++i)
      a[i] = load_fragA(xa + (size_t)i * 16 * Dc, Dc, k0, lane);
#pragma unroll
    for (int j = 0; j < 2; ++j)
      b[j] = load_fragB(wb + (size_t)j * 16 * Dc, Dc, k0, lane);
#pragma unroll
    for (int i = 0; i < 4; ++i)
#pragma unroll
      for (int j = 0; j < 2; ++j) c[i][j] = wmma_bf16(a[i], b[j], c[i][j]);
  }

#pragma unroll
  for (int i = 0; i < 4; ++i)
#pragma unroll
    for (int j = 0; j < 2; ++j)
#pragma unroll
      for (int r = 0; r < 8; ++r) {
        const int m = m_base + i * 16 + ((lane >> 4) << 3) + r;
        const int n = n_base + j * 16 + (lane & 15);
        const float v = c[i][j][r] + bias[n];
        if (MODE == 2) {
          ((float*)outp)[(size_t)m * Dc + n] = v;
        } else {
          const int bb = m >> 11, s = m & (Sc - 1);
          const int h = n >> 6,  d = n & (HDc - 1);
          const __bf16 bvv = f32_to_bf16(v);
          if (MODE == 0)
            ((__bf16*)outp)[(((size_t)bb * Hc + h) * Sc + s) * HDc + d] = bvv;
          else
            ((__bf16*)outp)[(((size_t)bb * Hc + h) * HDc + d) * Sc + s] = bvv;
        }
      }
}

// ------------------------------------------------------- flash attention
// Q,K: [B][H][S][HD] bf16; Vt: [B][H][HD][S] bf16; ctx out: [B][S][D] bf16.
// One wave handles 32 q-rows for one (b,h); online softmax over 64-key blocks
// (wide blocks amortize the shuffle-reduction / rescale cost per key).
__global__ __launch_bounds__(128) void attn_kernel(
    const __bf16* __restrict__ Q, const __bf16* __restrict__ K,
    const __bf16* __restrict__ Vt, __bf16* __restrict__ ctx) {
  __shared__ __bf16 smemP[4][32 * 64];

  const int lane = threadIdx.x & 31;
  const int wid  = threadIdx.x >> 5;
  const int bh   = blockIdx.y;            // b*H + h
  const int b    = bh >> 4;
  const int h    = bh & (Hc - 1);
  const int q0   = (blockIdx.x * 4 + wid) * 32;

  // Q fragments: reused for the whole key loop
  const __bf16* Qb = Q + ((size_t)bh * Sc + q0) * HDc;
  v16bf aq[2][2];
#pragma unroll
  for (int i = 0; i < 2; ++i)
#pragma unroll
    for (int kk = 0; kk < 2; ++kk)
      aq[i][kk] = load_fragA(Qb + (size_t)i * 16 * HDc, HDc, kk * 32, lane);

  v8f acc[2][4];
#pragma unroll
  for (int i = 0; i < 2; ++i)
#pragma unroll
    for (int jj = 0; jj < 4; ++jj) acc[i][jj] = (v8f){};
  float mrow[2][8], lrow[2][8];
#pragma unroll
  for (int i = 0; i < 2; ++i)
#pragma unroll
    for (int r = 0; r < 8; ++r) { mrow[i][r] = -3.0e38f; lrow[i][r] = 0.0f; }

  const __bf16* Vb = Vt + (size_t)bh * HDc * Sc;

  for (int n0 = 0; n0 < Sc; n0 += 64) {
    const __bf16* Kb = K + ((size_t)bh * Sc + n0) * HDc;
    if (n0 + 64 < Sc) {
      __builtin_prefetch(Kb + 64 * HDc, 0, 3);   // next K tile
      __builtin_prefetch(Vb + n0 + 64, 0, 3);    // next V tile
    }

    // ---- scores S = Q * K^T  (32 q x 64 k), 16 WMMAs
    v8f sc[2][4];
#pragma unroll
    for (int j = 0; j < 4; ++j) {
      v16bf bk0 = load_fragB(Kb + (size_t)j * 16 * HDc, HDc, 0, lane);
      v16bf bk1 = load_fragB(Kb + (size_t)j * 16 * HDc, HDc, 32, lane);
#pragma unroll
      for (int i = 0; i < 2; ++i) {
        v8f t = (v8f){};
        t = wmma_bf16(aq[i][0], bk0, t);
        t = wmma_bf16(aq[i][1], bk1, t);
        sc[i][j] = t;
      }
    }

    // ---- online softmax (row stats live on 16-lane halves per C layout)
#pragma unroll
    for (int i = 0; i < 2; ++i)
#pragma unroll
      for (int r = 0; r < 8; ++r) {
        float t = fmaxf(fmaxf(sc[i][0][r], sc[i][1][r]),
                        fmaxf(sc[i][2][r], sc[i][3][r]));
#pragma unroll
        for (int off = 1; off < 16; off <<= 1)
          t = fmaxf(t, __shfl_xor(t, off, 32));
        const float mnew = fmaxf(mrow[i][r], t);
        const float corr = __builtin_amdgcn_exp2f((mrow[i][r] - mnew) * kC1);
        float rs = 0.0f;
#pragma unroll
        for (int j = 0; j < 4; ++j) {
          const float p = __builtin_amdgcn_exp2f((sc[i][j][r] - mnew) * kC1);
          sc[i][j][r] = p;
          rs += p;
        }
#pragma unroll
        for (int off = 1; off < 16; off <<= 1) rs += __shfl_xor(rs, off, 32);
        lrow[i][r] = lrow[i][r] * corr + rs;
        mrow[i][r] = mnew;
#pragma unroll
        for (int jj = 0; jj < 4; ++jj) acc[i][jj][r] *= corr;
      }

    // ---- P (D-layout f32) -> LDS bf16 (32x64) -> A-fragment re-layout
#pragma unroll
    for (int i = 0; i < 2; ++i)
#pragma unroll
      for (int j = 0; j < 4; ++j)
#pragma unroll
        for (int r = 0; r < 8; ++r) {
          const int prow = i * 16 + ((lane >> 4) << 3) + r;
          const int pcol = j * 16 + (lane & 15);
          smemP[wid][prow * 64 + pcol] = f32_to_bf16(sc[i][j][r]);
        }
    asm volatile("s_wait_dscnt 0" ::: "memory");

    v16bf ap[2][2];
#pragma unroll
    for (int i = 0; i < 2; ++i)
#pragma unroll
      for (int kk = 0; kk < 2; ++kk) {
        const int prow = i * 16 + (lane & 15);
        const int pk   = kk * 32 + ((lane >> 4) << 3);
        const __bf16* pp = &smemP[wid][prow * 64 + pk];
        BfPair pr{*(const v8bf*)(pp), *(const v8bf*)(pp + 16)};
        ap[i][kk] = __builtin_bit_cast(v16bf, pr);
      }

    // ---- acc += P(32x64) * V(64x64), 16 WMMAs (Vt rows contiguous in keys)
#pragma unroll
    for (int jj = 0; jj < 4; ++jj) {
      v16bf bv0 = load_fragB(Vb + (size_t)jj * 16 * Sc, Sc, n0, lane);
      v16bf bv1 = load_fragB(Vb + (size_t)jj * 16 * Sc, Sc, n0 + 32, lane);
#pragma unroll
      for (int i = 0; i < 2; ++i) {
        acc[i][jj] = wmma_bf16(ap[i][0], bv0, acc[i][jj]);
        acc[i][jj] = wmma_bf16(ap[i][1], bv1, acc[i][jj]);
      }
    }
  }

  // ---- epilogue: divide by row sum, store ctx[B][S][D] bf16
#pragma unroll
  for (int i = 0; i < 2; ++i)
#pragma unroll
    for (int r = 0; r < 8; ++r) {
      const float inv = 1.0f / lrow[i][r];
      const int s = q0 + i * 16 + ((lane >> 4) << 3) + r;
#pragma unroll
      for (int jj = 0; jj < 4; ++jj) {
        const int d = jj * 16 + (lane & 15);
        ctx[((size_t)b * Sc + s) * Dc + h * HDc + d] =
            f32_to_bf16(acc[i][jj][r] * inv);
      }
    }
}

// ---------------------------------------------------------------- launcher
extern "C" void kernel_launch(void* const* d_in, const int* in_sizes, int n_in,
                              void* d_out, int out_size, void* d_ws,
                              size_t ws_size, hipStream_t stream) {
  const float* x  = (const float*)d_in[0];
  const float* Wq = (const float*)d_in[1];
  const float* bq = (const float*)d_in[2];
  const float* Wk = (const float*)d_in[3];
  const float* bk = (const float*)d_in[4];
  const float* Wv = (const float*)d_in[5];
  const float* bv = (const float*)d_in[6];
  const float* Wo = (const float*)d_in[7];
  const float* bo = (const float*)d_in[8];

  // workspace layout (48 MiB total, bf16)
  __bf16* xb  = (__bf16*)d_ws;                 // [4096][1024]
  __bf16* wqb = xb  + (size_t)SBc * Dc;        // [1024][1024]
  __bf16* wkb = wqb + (size_t)Dc * Dc;
  __bf16* wvb = wkb + (size_t)Dc * Dc;
  __bf16* wob = wvb + (size_t)Dc * Dc;
  __bf16* Qh  = wob + (size_t)Dc * Dc;         // [B][H][S][HD]
  __bf16* Kh  = Qh + (size_t)SBc * Dc;         // [B][H][S][HD]
  __bf16* Vt  = Kh + (size_t)SBc * Dc;         // [B][H][HD][S]
  __bf16* ctx = Vt + (size_t)SBc * Dc;         // [B][S][D]

  cvt_f32_bf16<<<512, 256, 0, stream>>>(x,  xb,  SBc * Dc);
  cvt_f32_bf16<<<256, 256, 0, stream>>>(Wq, wqb, Dc * Dc);
  cvt_f32_bf16<<<256, 256, 0, stream>>>(Wk, wkb, Dc * Dc);
  cvt_f32_bf16<<<256, 256, 0, stream>>>(Wv, wvb, Dc * Dc);
  cvt_f32_bf16<<<256, 256, 0, stream>>>(Wo, wob, Dc * Dc);

  dim3 gg(SBc / 64, Dc / 256);
  gemm_bf16<0><<<gg, 256, 0, stream>>>(xb, wqb, bq, Qh);
  gemm_bf16<0><<<gg, 256, 0, stream>>>(xb, wkb, bk, Kh);
  gemm_bf16<1><<<gg, 256, 0, stream>>>(xb, wvb, bv, Vt);

  attn_kernel<<<dim3(Sc / 128, Bc * Hc), 128, 0, stream>>>(Qh, Kh, Vt, ctx);

  gemm_bf16<2><<<gg, 256, 0, stream>>>(ctx, wob, bo, d_out);
}
